// SCENE15_JointMembership_89816356094172
// MI455X (gfx1250) — matrix operand order repaired and checked by reading, not compiled, not run on hardware
//
#include <hip/hip_runtime.h>
#include <hip/hip_bf16.h>

typedef __attribute__((ext_vector_type(16))) _Float16 v16h;
typedef __attribute__((ext_vector_type(8)))  _Float16 v8h;
typedef __attribute__((ext_vector_type(8)))  float    v8f;

#define B_N   16384   // batch
#define IN_D  200     // input features
#define IN_DP 224     // padded to multiple of 32 (7 WMMA K-steps)
#define HID_  256
#define NCLS  15
#define NCLSP 16
#define NLAY  3

// ---------------------------------------------------------------------------
// f32 -> f16 convert with zero padding (rows and cols). dst is [Rp, Cp].
// ---------------------------------------------------------------------------
__global__ void cvt_pad_f16(const float* __restrict__ src, _Float16* __restrict__ dst,
                            int R, int C, int Rp, int Cp) {
    int idx = blockIdx.x * blockDim.x + threadIdx.x;
    int total = Rp * Cp;
    if (idx >= total) return;
    int r = idx / Cp, c = idx - r * Cp;
    float v = (r < R && c < C) ? src[r * C + c] : 0.0f;
    dst[idx] = (_Float16)v;
}

// ---------------------------------------------------------------------------
// Quantum branch: one thread per (b, k). The per-layer trainable gates
// RZ(t2)·RY(t1)·RZ(t0) are folded into one SU(2) matrix [[A,B],[-B*,A*]]
// with A = q·p·cos(t1/2), B = -q·conj(p)·sin(t1/2), p=e^{-i t0/2}, q=e^{-i t2/2}.
// State stays normalized, so (<Z>+1)/2 == |s0|^2 and
// fuzzy[b][k] = prod_f |s0(b,f,k)|^2.
// ---------------------------------------------------------------------------
__global__ void quantum_fuzzy_kernel(const float* __restrict__ x,
                                     const float* __restrict__ theta,
                                     float* __restrict__ fuzzy) {
    int idx = blockIdx.x * blockDim.x + threadIdx.x;
    if (idx >= B_N * NCLS) return;
    int b = idx / NCLS;
    int k = idx - b * NCLS;

    float Ar[NLAY], Ai[NLAY], Br[NLAY], Bi[NLAY];
#pragma unroll
    for (int l = 0; l < NLAY; ++l) {
        float t0 = theta[(k * NLAY + l) * 3 + 0];
        float t1 = theta[(k * NLAY + l) * 3 + 1];
        float t2 = theta[(k * NLAY + l) * 3 + 2];
        float s0_, c0_, s1_, c1_, s2_, c2_;
        __sincosf(0.5f * t0, &s0_, &c0_);   // p = (c0, -s0)
        __sincosf(0.5f * t1, &s1_, &c1_);   // cb = c1, sb = s1
        __sincosf(0.5f * t2, &s2_, &c2_);   // q = (c2, -s2)
        float qpr = c2_ * c0_ - s2_ * s0_;            // q*p
        float qpi = -(c2_ * s0_ + s2_ * c0_);
        float qcr = c2_ * c0_ + s2_ * s0_;            // q*conj(p)
        float qci = c2_ * s0_ - s2_ * c0_;
        Ar[l] = qpr * c1_;  Ai[l] = qpi * c1_;
        Br[l] = -qcr * s1_; Bi[l] = -qci * s1_;
    }

    const float* xr = x + (long)b * IN_D;
    float acc = 1.0f;
    for (int f = 0; f < IN_D; ++f) {
        float a  = 0.5f * xr[f];
        float cx = __cosf(a), sx = __sinf(a);
        float s0r = 1.0f, s0i = 0.0f, s1r = 0.0f, s1i = 0.0f;
#pragma unroll
        for (int l = 0; l < NLAY; ++l) {
            // data encoder RY(x)
            float n0r = cx * s0r - sx * s1r, n0i = cx * s0i - sx * s1i;
            float n1r = sx * s0r + cx * s1r, n1i = sx * s0i + cx * s1i;
            // fused trainable SU(2): s0 = A n0 + B n1 ; s1 = -B* n0 + A* n1
            s0r = Ar[l] * n0r - Ai[l] * n0i + Br[l] * n1r - Bi[l] * n1i;
            s0i = Ar[l] * n0i + Ai[l] * n0r + Br[l] * n1i + Bi[l] * n1r;
            s1r = -Br[l] * n0r - Bi[l] * n0i + Ar[l] * n1r + Ai[l] * n1i;
            s1i =  Bi[l] * n0r - Br[l] * n0i + Ar[l] * n1i - Ai[l] * n1r;
        }
        acc *= (s0r * s0r + s0i * s0i);
    }
    fuzzy[idx] = acc;
}

// ---------------------------------------------------------------------------
// fusion = c3 + fuzzy @ Wf^T + bf   (output f16 for next WMMA GEMM)
// ---------------------------------------------------------------------------
__global__ void fusion_kernel(const _Float16* __restrict__ c3,
                              const float* __restrict__ fuzzy,
                              const float* __restrict__ Wf,
                              const float* __restrict__ bf,
                              _Float16* __restrict__ out) {
    int idx = blockIdx.x * blockDim.x + threadIdx.x;
    if (idx >= B_N * HID_) return;
    int b = idx >> 8;
    int h = idx & 255;
    float s = (float)c3[idx] + bf[h];
    const float* fz = fuzzy + b * NCLS;
    const float* w  = Wf + h * NCLS;
#pragma unroll
    for (int k = 0; k < NCLS; ++k) s += fz[k] * w[k];
    out[idx] = (_Float16)s;
}

// ---------------------------------------------------------------------------
// WMMA GEMM: C[M, N] = act(A[M,Kd] @ Wt[Npad,Kd]^T + bias)
// Block: 256 threads = 8 waves. Tile: BM=128 (16 rows/wave), BN=NT*16.
// The full B-panel (NT*16 x Kd f16, <=32KB of the 320KB WGP LDS) is staged
// once; the K-loop then runs barrier-free. A sched_barrier pins all NT
// B-fragment ds_loads ahead of the WMMA chain so a single s_wait_dscnt
// covers NT back-to-back v_wmma issues (16-bit WMMA co-executes as TRANS).
// A and B fragments both use contiguous K per lane half (half h: K 16h..16h+15).
// ---------------------------------------------------------------------------
template <int NT, bool RELU, bool STORE_F16, bool STORE_F32>
__global__ __launch_bounds__(256)
void wmma_gemm_kernel(const _Float16* __restrict__ A,   // [M, Kd]
                      const _Float16* __restrict__ Wt,  // [Npad, Kd]
                      const float* __restrict__ bias,   // [N]
                      _Float16* __restrict__ Ch,        // [M, ldc16]
                      float* __restrict__ Cf,           // [M, ldcf]
                      int M, int N, int Npad, int Kd, int ldc16, int ldcf) {
    __shared__ __align__(32) _Float16 Bs[NT * 16 * HID_];  // Kd <= HID_

    const int tid  = threadIdx.x;
    const int wave = tid >> 5;
    const int lane = tid & 31;
    const int half = lane >> 4;    // which 16-lane group
    const int l16  = lane & 15;

    const int rowBase = blockIdx.x * 128 + wave * 16;
    const int colBase = blockIdx.y * (NT * 16);

    // ---- stage entire B panel in LDS (once) -------------------------------
    {
        const int chunksPerRow = Kd >> 3;                 // 16B chunks per row
        const int totalChunks  = NT * 16 * chunksPerRow;
        for (int i = tid; i < totalChunks; i += 256) {
            int n  = i / chunksPerRow;
            int kk = (i - n * chunksPerRow) << 3;
            int gn = colBase + n;
            v8h bv = {};
            if (gn < Npad) bv = *(const v8h*)(Wt + (long)gn * Kd + kk);
            *(v8h*)(&Bs[n * Kd + kk]) = bv;
        }
    }
    __syncthreads();

    const _Float16* Arow = A + (long)(rowBase + l16) * Kd;

    v8f acc[NT] = {};

    // ---- barrier-free K loop ----------------------------------------------
    for (int k0 = 0; k0 < Kd; k0 += 32) {
        // A fragment: lane half 0 holds K {0..15}; half 1 holds {16..31}
        union { v16h v; v8h h[2]; } au;
        au.h[0] = *(const v8h*)(Arow + k0 + half * 16);
        au.h[1] = *(const v8h*)(Arow + k0 + half * 16 + 8);
        __builtin_prefetch(Arow + k0 + 32, 0, 1);   // global_prefetch_b8

        // gather all B fragments into distinct live registers
        v16h bfrag[NT];
#pragma unroll
        for (int nt = 0; nt < NT; ++nt) {
            // B fragment: lane = column (l16), K split 0..15 / 16..31 per lane half
            bfrag[nt] = *(const v16h*)(&Bs[(nt * 16 + l16) * Kd + k0 + half * 16]);
        }
        // hard scheduling fence: loads stay above, WMMAs stay below, so one
        // dscnt wait covers an uninterrupted NT-deep v_wmma burst.
        __builtin_amdgcn_sched_barrier(0);
#pragma unroll
        for (int nt = 0; nt < NT; ++nt) {
            acc[nt] = __builtin_amdgcn_wmma_f32_16x16x32_f16(
                false, au.v, false, bfrag[nt], (short)0, acc[nt], false, false);
        }
    }

    // ---- epilogue: C/D layout — VGPR r: lanes 0-15 -> M=r, 16-31 -> M=8+r --
    union { v8f v; float f[8]; } cu;
#pragma unroll
    for (int nt = 0; nt < NT; ++nt) {
        const int col = colBase + nt * 16 + l16;
        const float bval = (col < N) ? bias[col] : 0.0f;
        cu.v = acc[nt];
#pragma unroll
        for (int r = 0; r < 8; ++r) {
            const int row = rowBase + r + half * 8;
            float v = cu.f[r] + bval;
            if (RELU) v = fmaxf(v, 0.0f);
            if (STORE_F16) Ch[(long)row * ldc16 + col] = (_Float16)v;
            if (STORE_F32) { if (col < N) Cf[(long)row * ldcf + col] = v; }
        }
    }
}

// ---------------------------------------------------------------------------
extern "C" void kernel_launch(void* const* d_in, const int* in_sizes, int n_in,
                              void* d_out, int out_size, void* d_ws, size_t ws_size,
                              hipStream_t stream) {
    (void)in_sizes; (void)n_in; (void)out_size; (void)ws_size;

    const float* x    = (const float*)d_in[0];
    const float* W1   = (const float*)d_in[1];
    const float* b1   = (const float*)d_in[2];
    const float* W2   = (const float*)d_in[3];
    const float* b2   = (const float*)d_in[4];
    const float* W3   = (const float*)d_in[5];
    const float* b3   = (const float*)d_in[6];
    const float* th   = (const float*)d_in[7];
    const float* Wf   = (const float*)d_in[8];
    const float* bf   = (const float*)d_in[9];
    const float* Wc1  = (const float*)d_in[10];
    const float* bc1  = (const float*)d_in[11];
    const float* Wc2  = (const float*)d_in[12];
    const float* bc2  = (const float*)d_in[13];
    const float* Wc3  = (const float*)d_in[14];
    const float* bc3  = (const float*)d_in[15];
    const float* Wc4  = (const float*)d_in[16];
    const float* bc4  = (const float*)d_in[17];
    float* out = (float*)d_out;

    char* p = (char*)d_ws;
    auto alloc = [&](size_t bytes) -> char* {
        char* r = p;
        p += (bytes + 255) & ~(size_t)255;
        return r;
    };
    _Float16* xh   = (_Float16*)alloc((size_t)B_N * IN_DP * 2);
    _Float16* W1h  = (_Float16*)alloc((size_t)HID_ * IN_DP * 2);
    _Float16* W2h  = (_Float16*)alloc((size_t)HID_ * HID_ * 2);
    _Float16* W3h  = (_Float16*)alloc((size_t)HID_ * HID_ * 2);
    _Float16* Wc1h = (_Float16*)alloc((size_t)HID_ * HID_ * 2);
    _Float16* Wc2h = (_Float16*)alloc((size_t)HID_ * HID_ * 2);
    _Float16* Wc3h = (_Float16*)alloc((size_t)HID_ * HID_ * 2);
    _Float16* Wc4h = (_Float16*)alloc((size_t)NCLSP * HID_ * 2);
    _Float16* bufA = (_Float16*)alloc((size_t)B_N * HID_ * 2);
    _Float16* bufB = (_Float16*)alloc((size_t)B_N * HID_ * 2);
    float*    fuzz = (float*)alloc((size_t)B_N * NCLS * 4);

    // ---- precision conversion / padding -----------------------------------
    {
        int t = B_N * IN_DP;
        cvt_pad_f16<<<(t + 255) / 256, 256, 0, stream>>>(x, xh, B_N, IN_D, B_N, IN_DP);
    }
    {
        int t = HID_ * IN_DP;
        cvt_pad_f16<<<(t + 255) / 256, 256, 0, stream>>>(W1, W1h, HID_, IN_D, HID_, IN_DP);
    }
    {
        int t = HID_ * HID_;
        int g = (t + 255) / 256;
        cvt_pad_f16<<<g, 256, 0, stream>>>(W2,  W2h,  HID_, HID_, HID_, HID_);
        cvt_pad_f16<<<g, 256, 0, stream>>>(W3,  W3h,  HID_, HID_, HID_, HID_);
        cvt_pad_f16<<<g, 256, 0, stream>>>(Wc1, Wc1h, HID_, HID_, HID_, HID_);
        cvt_pad_f16<<<g, 256, 0, stream>>>(Wc2, Wc2h, HID_, HID_, HID_, HID_);
        cvt_pad_f16<<<g, 256, 0, stream>>>(Wc3, Wc3h, HID_, HID_, HID_, HID_);
    }
    {
        int t = NCLSP * HID_;
        cvt_pad_f16<<<(t + 255) / 256, 256, 0, stream>>>(Wc4, Wc4h, NCLS, HID_, NCLSP, HID_);
    }

    // ---- quantum fuzzy branch (independent of the MLP; runs before it on one stream) ----------
    {
        int t = B_N * NCLS;
        quantum_fuzzy_kernel<<<(t + 255) / 256, 256, 0, stream>>>(x, th, fuzz);
    }

    // ---- classical MLP: 3x Linear+ReLU on WMMA ----------------------------
    const dim3 blk(256);
    const dim3 gHid(B_N / 128, HID_ / 64);
    wmma_gemm_kernel<4, true, true, false><<<gHid, blk, 0, stream>>>(
        xh, W1h, b1, bufA, nullptr, B_N, HID_, HID_, IN_DP, HID_, 0);
    wmma_gemm_kernel<4, true, true, false><<<gHid, blk, 0, stream>>>(
        bufA, W2h, b2, bufB, nullptr, B_N, HID_, HID_, HID_, HID_, 0);
    wmma_gemm_kernel<4, true, true, false><<<gHid, blk, 0, stream>>>(
        bufB, W3h, b3, bufA, nullptr, B_N, HID_, HID_, HID_, HID_, 0);

    // ---- ADD fusion with quantum branch -----------------------------------
    {
        int t = B_N * HID_;
        fusion_kernel<<<(t + 255) / 256, 256, 0, stream>>>(bufA, fuzz, Wf, bf, bufB);
    }

    // ---- classifier: 3x Linear+ReLU + final projection --------------------
    wmma_gemm_kernel<4, true, true, false><<<gHid, blk, 0, stream>>>(
        bufB, Wc1h, bc1, bufA, nullptr, B_N, HID_, HID_, HID_, HID_, 0);
    wmma_gemm_kernel<4, true, true, false><<<gHid, blk, 0, stream>>>(
        bufA, Wc2h, bc2, bufB, nullptr, B_N, HID_, HID_, HID_, HID_, 0);
    wmma_gemm_kernel<4, true, true, false><<<gHid, blk, 0, stream>>>(
        bufB, Wc3h, bc3, bufA, nullptr, B_N, HID_, HID_, HID_, HID_, 0);

    const dim3 gOut(B_N / 128, 1);
    wmma_gemm_kernel<1, false, false, true><<<gOut, blk, 0, stream>>>(
        bufA, Wc4h, bc4, nullptr, out, B_N, NCLS, NCLSP, HID_, 0, NCLS);
}
